// GraphNMFSimCLRNetworkCustom_36979668419190
// MI455X (gfx1250) — matrix operand order repaired and checked by reading, not compiled, not run on hardware
//
#include <hip/hip_runtime.h>
#include <hip/hip_bf16.h>

typedef __attribute__((ext_vector_type(16))) _Float16 v16h_t;
typedef __attribute__((ext_vector_type(8)))  _Float16 v8h_t;
typedef __attribute__((ext_vector_type(8)))  float    v8f_t;
typedef __attribute__((ext_vector_type(4)))  float    v4f_t;

#if defined(__has_builtin)
#if __has_builtin(__builtin_amdgcn_sched_barrier)
#define SCHED_BARRIER() __builtin_amdgcn_sched_barrier(0)
#endif
#endif
#ifndef SCHED_BARRIER
#define SCHED_BARRIER()
#endif

__device__ __forceinline__ float lrelu(float x) { return x >= 0.0f ? x : 0.01f * x; }

// ---------------- conv1d(valid, k=3, 4 out channels) + leaky, reshaped to N x 112 ----
__global__ void conv_kernel(const float* __restrict__ xr, const float* __restrict__ cw,
                            const float* __restrict__ cb, float* __restrict__ h, int N) {
    int idx = blockIdx.x * blockDim.x + threadIdx.x;
    if (idx >= N * 112) return;
    int n = idx / 112, r = idx - n * 112;
    int c = r / 28, p = r - c * 28;
    float acc = cb[c];
#pragma unroll
    for (int t = 0; t < 3; ++t) acc += cw[c * 3 + t] * xr[n * 30 + p + t];
    h[idx] = lrelu(acc);
}

// ---------------- row L2-normalize f32 -> f16, zero-padded K -> KP ----------------
__global__ void norm16_kernel(const float* __restrict__ x, _Float16* __restrict__ xn,
                              int N, int Kin, int KP) {
    int row  = blockIdx.x * (blockDim.x >> 5) + (threadIdx.x >> 5);
    int lane = threadIdx.x & 31;
    if (row >= N) return;
    float ss = 0.0f;
    for (int k = lane; k < Kin; k += 32) { float v = x[(size_t)row * Kin + k]; ss += v * v; }
#pragma unroll
    for (int off = 16; off >= 1; off >>= 1) ss += __shfl_xor(ss, off, 32);
    float inv = 1.0f / fmaxf(sqrtf(ss), 1e-12f);
    for (int k = lane; k < KP; k += 32)
        xn[(size_t)row * KP + k] =
            (k < Kin) ? (_Float16)(x[(size_t)row * Kin + k] * inv) : (_Float16)0.0f;
}

// ---------------- fused sim = xn @ xn^T (WMMA f16) + per-row partial top-6 -----------
// Wave: 32 rows (two 16-row M-tiles, A resident) x one quarter of the columns
// (split=4 -> 1024 waves, ~2/SIMD). Two column tiles per iteration: 16 b128 loads
// pinned before 16 WMMAs (sched_barrier) so latency is paid once per iteration.
// Next iteration's B rows prefetched during the scan phase.
template <int KP>
__global__ __launch_bounds__(128) void sim_topk_kernel(const _Float16* __restrict__ xn,
                                                       int* __restrict__ pidx,
                                                       float* __restrict__ pval, int N) {
    __shared__ float lds[4][4][16 * 20];   // [wave][tile nt*2+m][row*20+col], 16B rows
    const int wave  = threadIdx.x >> 5;
    const int lane  = threadIdx.x & 31;
    const int lhalf = lane >> 4;
    const int lrow  = lane & 15;
    const int gw    = blockIdx.x * 4 + wave;   // global wave id
    const int strip = gw >> 2;                 // 32-row strip
    const int split = gw & 3;                  // which column quarter
    const int r0    = strip * 32;
    const int cbeg  = split * (N >> 2);
    const int cend  = cbeg + (N >> 2);
    constexpr int NK = KP / 32;

    // A tiles (rows r0..r0+15 and r0+16..r0+31): per-lane K = {0..7,16..23} + 8*lhalf.
    v16h_t aa0[NK], aa1[NK];
#pragma unroll
    for (int kk = 0; kk < NK; ++kk) {
        const _Float16* ap0 = xn + (size_t)(r0 + lrow) * KP + lhalf * 8 + 32 * kk;
        const _Float16* ap1 = ap0 + (size_t)16 * KP;
        v8h_t lo0 = *(const v8h_t*)(ap0);
        v8h_t hi0 = *(const v8h_t*)(ap0 + 16);
        v8h_t lo1 = *(const v8h_t*)(ap1);
        v8h_t hi1 = *(const v8h_t*)(ap1 + 16);
        aa0[kk] = __builtin_shufflevector(lo0, hi0, 0, 1, 2, 3, 4, 5, 6, 7,
                                          8, 9, 10, 11, 12, 13, 14, 15);
        aa1[kk] = __builtin_shufflevector(lo1, hi1, 0, 1, 2, 3, 4, 5, 6, 7,
                                          8, 9, 10, 11, 12, 13, 14, 15);
    }

    float tv[6]; int ti[6];
#pragma unroll
    for (int q = 0; q < 6; ++q) { tv[q] = -3.0e38f; ti[q] = 0; }

    for (int c0 = cbeg; c0 < cend; c0 += 32) {
        // Stage B for two column tiles (c0 and c0+16): 16 b128 loads, all in flight
        // before the first WMMA (sched_barrier pins them).
        v16h_t bbA[NK], bbB[NK];
        const _Float16* bpA = xn + (size_t)(c0 + lrow) * KP + lhalf * 8;
        const _Float16* bpB = bpA + (size_t)16 * KP;
#pragma unroll
        for (int kk = 0; kk < NK; ++kk) {
            v8h_t loA = *(const v8h_t*)(bpA + 32 * kk);
            v8h_t hiA = *(const v8h_t*)(bpA + 32 * kk + 16);
            v8h_t loB = *(const v8h_t*)(bpB + 32 * kk);
            v8h_t hiB = *(const v8h_t*)(bpB + 32 * kk + 16);
            bbA[kk] = __builtin_shufflevector(loA, hiA, 0, 1, 2, 3, 4, 5, 6, 7,
                                              8, 9, 10, 11, 12, 13, 14, 15);
            bbB[kk] = __builtin_shufflevector(loB, hiB, 0, 1, 2, 3, 4, 5, 6, 7,
                                              8, 9, 10, 11, 12, 13, 14, 15);
        }
        SCHED_BARRIER();
        v8f_t cv00 = {0.f, 0.f, 0.f, 0.f, 0.f, 0.f, 0.f, 0.f};  // (M0, Ntile A)
        v8f_t cv01 = {0.f, 0.f, 0.f, 0.f, 0.f, 0.f, 0.f, 0.f};  // (M1, Ntile A)
        v8f_t cv10 = {0.f, 0.f, 0.f, 0.f, 0.f, 0.f, 0.f, 0.f};  // (M0, Ntile B)
        v8f_t cv11 = {0.f, 0.f, 0.f, 0.f, 0.f, 0.f, 0.f, 0.f};  // (M1, Ntile B)
#pragma unroll
        for (int kk = 0; kk < NK; ++kk) {   // each B fragment feeds two WMMAs
            cv00 = __builtin_amdgcn_wmma_f32_16x16x32_f16(false, aa0[kk], false, bbA[kk],
                                                          (short)0, cv00, false, false);
            cv01 = __builtin_amdgcn_wmma_f32_16x16x32_f16(false, aa1[kk], false, bbA[kk],
                                                          (short)0, cv01, false, false);
            cv10 = __builtin_amdgcn_wmma_f32_16x16x32_f16(false, aa0[kk], false, bbB[kk],
                                                          (short)0, cv10, false, false);
            cv11 = __builtin_amdgcn_wmma_f32_16x16x32_f16(false, aa1[kk], false, bbB[kk],
                                                          (short)0, cv11, false, false);
        }
        // Spill 4 C tiles to per-wave LDS (transpose lanes->rows).
#pragma unroll
        for (int v = 0; v < 8; ++v) {
            int mrow = (v + 8 * lhalf) * 20 + lrow;
            lds[wave][0][mrow] = cv00[v];
            lds[wave][1][mrow] = cv01[v];
            lds[wave][2][mrow] = cv10[v];
            lds[wave][3][mrow] = cv11[v];
        }
        __builtin_amdgcn_wave_barrier();
        asm volatile("" ::: "memory");
        // Prefetch next iteration's B rows (global_prefetch_b8) to overlap the scan.
        {
            int cp = (c0 + 32 < cend) ? (c0 + 32) : cbeg;
            const _Float16* pfA = xn + (size_t)(cp + lrow) * KP + lhalf * 16;
            __builtin_prefetch(pfA, 0, 3);
            __builtin_prefetch(pfA + (size_t)16 * KP, 0, 3);
        }
        // Each lane owns (m-tile = lhalf, row = lrow); scans both column tiles.
#pragma unroll
        for (int nt = 0; nt < 2; ++nt) {
            const float* rowp = &lds[wave][nt * 2 + lhalf][lrow * 20];
            v4f_t q0 = *(const v4f_t*)(rowp + 0);
            v4f_t q1 = *(const v4f_t*)(rowp + 4);
            v4f_t q2 = *(const v4f_t*)(rowp + 8);
            v4f_t q3 = *(const v4f_t*)(rowp + 12);
            float vals[16];
#pragma unroll
            for (int j = 0; j < 4; ++j) {
                vals[j] = q0[j]; vals[4 + j] = q1[j];
                vals[8 + j] = q2[j]; vals[12 + j] = q3[j];
            }
            float m = vals[0];
#pragma unroll
            for (int n = 1; n < 16; ++n) m = fmaxf(m, vals[n]);
            if (m > tv[5]) {               // rare after warm-up
                int cb = c0 + nt * 16;
#pragma unroll
                for (int n = 0; n < 16; ++n) {
                    float s = vals[n];
                    if (s > tv[5]) {
                        tv[5] = s; ti[5] = cb + n;
#pragma unroll
                        for (int q = 5; q >= 1; --q) {
                            if (tv[q] > tv[q - 1]) {
                                float tf = tv[q]; tv[q] = tv[q - 1]; tv[q - 1] = tf;
                                int tn = ti[q]; ti[q] = ti[q - 1]; ti[q - 1] = tn;
                            }
                        }
                    }
                }
            }
        }
        __builtin_amdgcn_wave_barrier();
        asm volatile("" ::: "memory");
    }
    // Emit partial (per column-quarter) sorted top-6 for this lane's row.
    int row = r0 + lhalf * 16 + lrow;
    size_t base = ((size_t)split * N + row) * 6;
#pragma unroll
    for (int q = 0; q < 6; ++q) { pidx[base + q] = ti[q]; pval[base + q] = tv[q]; }
}

// ---------------- merge the four sorted partial top-6 lists per row ------------------
__global__ void topk_merge_kernel(const int* __restrict__ pidx, const float* __restrict__ pval,
                                  int* __restrict__ tidx, float* __restrict__ tval, int N) {
    int row = blockIdx.x * blockDim.x + threadIdx.x;
    if (row >= N) return;
    int a0 = 0, a1 = 0, a2 = 0, a3 = 0;
    const size_t b0 = (size_t)row * 6;
    const size_t b1 = ((size_t)N + row) * 6;
    const size_t b2 = ((size_t)2 * N + row) * 6;
    const size_t b3 = ((size_t)3 * N + row) * 6;
#pragma unroll
    for (int q = 0; q < 6; ++q) {
        float f0 = (a0 < 6) ? pval[b0 + a0] : -3.0e38f;
        float f1 = (a1 < 6) ? pval[b1 + a1] : -3.0e38f;
        float f2 = (a2 < 6) ? pval[b2 + a2] : -3.0e38f;
        float f3 = (a3 < 6) ? pval[b3 + a3] : -3.0e38f;
        float best = f0; int which = 0;
        if (f1 > best) { best = f1; which = 1; }
        if (f2 > best) { best = f2; which = 2; }
        if (f3 > best) { best = f3; which = 3; }
        int bi;
        if (which == 0)      { bi = pidx[b0 + a0]; ++a0; }
        else if (which == 1) { bi = pidx[b1 + a1]; ++a1; }
        else if (which == 2) { bi = pidx[b2 + a2]; ++a2; }
        else                 { bi = pidx[b3 + a3]; ++a3; }
        tval[(size_t)row * 6 + q] = best;
        tidx[(size_t)row * 6 + q] = bi;
    }
}

// ---------------- graph utilities (sparse symmetric adjacency via scatter) -----------
__global__ void fill_kernel(float* __restrict__ p, float v, int n) {
    int i = blockIdx.x * blockDim.x + threadIdx.x;
    if (i < n) p[i] = v;
}

__global__ void degree_kernel(const int* __restrict__ idx, const float* __restrict__ val,
                              float* __restrict__ deg, int N) {
    int e = blockIdx.x * blockDim.x + threadIdx.x;
    if (e >= N * 6) return;
    int i = e / 6, j = idx[e];
    float v = val[e];
    if (i == j) {
        atomicAdd(&deg[i], v);
    } else {
        atomicAdd(&deg[i], 0.5f * v);
        atomicAdd(&deg[j], 0.5f * v);
    }
}

// Y[N,M] = X[N,K] @ W[K,M]
__global__ void linear_kernel(const float* __restrict__ X, const float* __restrict__ W,
                              float* __restrict__ Y, int N, int K, int M) {
    int idx = blockIdx.x * blockDim.x + threadIdx.x;
    if (idx >= N * M) return;
    int n = idx / M, m = idx - n * M;
    float acc = 0.0f;
    for (int k = 0; k < K; ++k) acc += X[(size_t)n * K + k] * W[k * M + m];
    Y[idx] = acc;
}

// z = D^-1/2 * y ; agg initialized to z (identity term of A_sym + I)
__global__ void scale_init_kernel(const float* __restrict__ deg, const float* __restrict__ y,
                                  float* __restrict__ z, float* __restrict__ agg, int N, int F) {
    int idx = blockIdx.x * blockDim.x + threadIdx.x;
    if (idx >= N * F) return;
    int i = idx / F;
    float zz = rsqrtf(deg[i]) * y[idx];
    z[idx] = zz;
    agg[idx] = zz;
}

__global__ void aggregate_kernel(const int* __restrict__ idx, const float* __restrict__ val,
                                 const float* __restrict__ z, float* __restrict__ agg,
                                 int N, int F) {
    int t = blockIdx.x * blockDim.x + threadIdx.x;
    if (t >= N * 6 * F) return;
    int e = t / F, f = t - e * F;
    int i = e / 6, j = idx[e];
    float v = val[e];
    if (i == j) {
        atomicAdd(&agg[(size_t)i * F + f], v * z[(size_t)j * F + f]);
    } else {
        atomicAdd(&agg[(size_t)i * F + f], 0.5f * v * z[(size_t)j * F + f]);
        atomicAdd(&agg[(size_t)j * F + f], 0.5f * v * z[(size_t)i * F + f]);
    }
}

__global__ void finalize_kernel(const float* __restrict__ deg, const float* __restrict__ agg,
                                const float* __restrict__ b, float* __restrict__ out,
                                int N, int F) {
    int idx = blockIdx.x * blockDim.x + threadIdx.x;
    if (idx >= N * F) return;
    int i = idx / F, f = idx - i * F;
    out[idx] = lrelu(rsqrtf(deg[i]) * agg[idx] + b[f]);
}

// ---------------- head: softmax over singleton axis == 1.0, so only Wv path matters --
__global__ void head_kernel(const float* __restrict__ h_raw, const float* __restrict__ in_w,
                            const float* __restrict__ in_b, const float* __restrict__ out_w,
                            const float* __restrict__ out_b, const float* __restrict__ cls_w,
                            const float* __restrict__ cls_b, float* __restrict__ out, int N) {
    int i = blockIdx.x * blockDim.x + threadIdx.x;
    if (i >= N) return;
    float hr[8], v[8], hc[8];
#pragma unroll
    for (int c = 0; c < 8; ++c) hr[c] = h_raw[(size_t)i * 8 + c];
#pragma unroll
    for (int r = 0; r < 8; ++r) {      // v = Wv @ hr + bv  (Wv = in_proj_w rows 16..23)
        float acc = in_b[16 + r];
#pragma unroll
        for (int c = 0; c < 8; ++c) acc += in_w[(16 + r) * 8 + c] * hr[c];
        v[r] = acc;
    }
#pragma unroll
    for (int r = 0; r < 8; ++r) {      // h_cross = Wout @ v + bout
        float acc = out_b[r];
#pragma unroll
        for (int c = 0; c < 8; ++c) acc += out_w[r * 8 + c] * v[c];
        hc[r] = acc;
    }
#pragma unroll
    for (int o = 0; o < 2; ++o) {      // cls over concat([hr, hc])
        float acc = cls_b[o];
#pragma unroll
        for (int c = 0; c < 8; ++c) acc += cls_w[o * 16 + c] * hr[c];
#pragma unroll
        for (int c = 0; c < 8; ++c) acc += cls_w[o * 16 + 8 + c] * hc[c];
        out[(size_t)i * 2 + o] = acc;
    }
}

extern "C" void kernel_launch(void* const* d_in, const int* in_sizes, int n_in,
                              void* d_out, int out_size, void* d_ws, size_t ws_size,
                              hipStream_t stream) {
    (void)n_in; (void)out_size; (void)ws_size;
    const float* x_raw  = (const float*)d_in[0];
    const float* conv_w = (const float*)d_in[2];
    const float* conv_b = (const float*)d_in[3];
    const float* rw1    = (const float*)d_in[4];
    const float* rb1    = (const float*)d_in[5];
    const float* rw2    = (const float*)d_in[6];
    const float* rb2    = (const float*)d_in[7];
    const float* in_w   = (const float*)d_in[12];
    const float* in_b   = (const float*)d_in[13];
    const float* out_w  = (const float*)d_in[14];
    const float* out_b  = (const float*)d_in[15];
    const float* cls_w  = (const float*)d_in[22];
    const float* cls_b  = (const float*)d_in[23];

    const int N = in_sizes[0] / 30;     // 8192

    char* p = (char*)d_ws;
    auto carve = [&](size_t bytes) -> char* {
        char* r = p; p += (bytes + 255) & ~(size_t)255; return r;
    };
    float*    h    = (float*)   carve((size_t)N * 112 * sizeof(float));
    _Float16* hn   = (_Float16*)carve((size_t)N * 128 * sizeof(_Float16));
    int*      pid  = (int*)     carve((size_t)4 * N * 6 * sizeof(int));
    float*    pvl  = (float*)   carve((size_t)4 * N * 6 * sizeof(float));
    int*      tidx = (int*)     carve((size_t)N * 6 * sizeof(int));
    float*    tval = (float*)   carve((size_t)N * 6 * sizeof(float));
    float*    deg  = (float*)   carve((size_t)N * sizeof(float));
    float*    y    = (float*)   carve((size_t)N * 32 * sizeof(float));
    float*    z    = (float*)   carve((size_t)N * 32 * sizeof(float));
    float*    agg  = (float*)   carve((size_t)N * 32 * sizeof(float));
    float*    h1   = (float*)   carve((size_t)N * 32 * sizeof(float));
    float*    h2   = (float*)   carve((size_t)N * 8 * sizeof(float));

    const int T = 256;
    conv_kernel<<<(N * 112 + T - 1) / T, T, 0, stream>>>(x_raw, conv_w, conv_b, h, N);
    norm16_kernel<<<(N + 7) / 8, 256, 0, stream>>>(h, hn, N, 112, 128);
    // waves = (N/32 strips) * 4 column-quarters; 4 waves per 128-thread block
    sim_topk_kernel<128><<<N / 32, 128, 0, stream>>>(hn, pid, pvl, N);
    topk_merge_kernel<<<(N + T - 1) / T, T, 0, stream>>>(pid, pvl, tidx, tval, N);

    fill_kernel<<<(N + T - 1) / T, T, 0, stream>>>(deg, 1.0f, N);
    degree_kernel<<<(N * 6 + T - 1) / T, T, 0, stream>>>(tidx, tval, deg, N);

    // GCN layer 1: 112 -> 32
    linear_kernel<<<(N * 32 + T - 1) / T, T, 0, stream>>>(h, rw1, y, N, 112, 32);
    scale_init_kernel<<<(N * 32 + T - 1) / T, T, 0, stream>>>(deg, y, z, agg, N, 32);
    aggregate_kernel<<<(N * 6 * 32 + T - 1) / T, T, 0, stream>>>(tidx, tval, z, agg, N, 32);
    finalize_kernel<<<(N * 32 + T - 1) / T, T, 0, stream>>>(deg, agg, rb1, h1, N, 32);

    // GCN layer 2: 32 -> 8
    linear_kernel<<<(N * 8 + T - 1) / T, T, 0, stream>>>(h1, rw2, y, N, 32, 8);
    scale_init_kernel<<<(N * 8 + T - 1) / T, T, 0, stream>>>(deg, y, z, agg, N, 8);
    aggregate_kernel<<<(N * 6 * 8 + T - 1) / T, T, 0, stream>>>(tidx, tval, z, agg, N, 8);
    finalize_kernel<<<(N * 8 + T - 1) / T, T, 0, stream>>>(deg, agg, rb2, h2, N, 8);

    head_kernel<<<(N + T - 1) / T, T, 0, stream>>>(h2, in_w, in_b, out_w, out_b,
                                                   cls_w, cls_b, (float*)d_out, N);
}
